// GatedDeltaNet_32521492365783
// MI455X (gfx1250) — matrix operand order repaired and checked by reading, not compiled, not run on hardware
//
#include <hip/hip_runtime.h>
#include <hip/hip_bf16.h>

typedef __attribute__((ext_vector_type(16))) __bf16 v16bf;
typedef __attribute__((ext_vector_type(8)))  __bf16 v8bf;
typedef __attribute__((ext_vector_type(8)))  float  v8f;

#define B_   2
#define S_   4096
#define H_   1024
#define NKH_ 8
#define NVH_ 16
#define DK_  64
#define DV_  64
#define KEY_DIM_  512
#define VAL_DIM_  1024
#define CONV_DIM_ 2048

// ---------------------------------------------------------------------------
// WMMA GEMM: C[M,N] = A[M,K] * W[N,K]^T  (f32 in, bf16 WMMA f32-accum, f32 out)
// Block = 128 threads (4 waves). Block tile 64x64, wave tile 32x32, BK = 32.
// Double-buffered LDS; next K-tile staged in registers while WMMAs run.
// ---------------------------------------------------------------------------
__global__ __launch_bounds__(128)
void gdn_wmma_gemm_xWT(const float* __restrict__ A, const float* __restrict__ W,
                       float* __restrict__ C, int M, int N, int K) {
    __shared__ alignas(64) __bf16 As[2][64][32];
    __shared__ alignas(64) __bf16 Bs[2][64][32];

    const int tid  = threadIdx.x;
    const int wave = tid >> 5;
    const int lane = tid & 31;
    const int tileM = blockIdx.y * 64;
    const int tileN = blockIdx.x * 64;
    const int wm = (wave >> 1) * 32;
    const int wn = (wave & 1) * 32;

    // staging: each thread owns 16 contiguous floats of one row (64 rows x 2 halves)
    const int srow = tid >> 1;
    const int skc  = (tid & 1) * 16;
    const float* Arow = &A[(size_t)(tileM + srow) * K + skc];
    const float* Wrow = &W[(size_t)(tileN + srow) * K + skc];

    v8f acc[2][2] = {};

    // ---- prologue: stage tile k0=0 into buffer 0 ----
    {
        float4 ra[4], rw[4];
        #pragma unroll
        for (int i = 0; i < 4; ++i) ra[i] = *(const float4*)(Arow + i * 4);
        #pragma unroll
        for (int i = 0; i < 4; ++i) rw[i] = *(const float4*)(Wrow + i * 4);
        v8bf pa[2], pw[2];
        #pragma unroll
        for (int i = 0; i < 4; ++i) {
            pa[i >> 1][(i & 1) * 4 + 0] = (__bf16)ra[i].x;
            pa[i >> 1][(i & 1) * 4 + 1] = (__bf16)ra[i].y;
            pa[i >> 1][(i & 1) * 4 + 2] = (__bf16)ra[i].z;
            pa[i >> 1][(i & 1) * 4 + 3] = (__bf16)ra[i].w;
            pw[i >> 1][(i & 1) * 4 + 0] = (__bf16)rw[i].x;
            pw[i >> 1][(i & 1) * 4 + 1] = (__bf16)rw[i].y;
            pw[i >> 1][(i & 1) * 4 + 2] = (__bf16)rw[i].z;
            pw[i >> 1][(i & 1) * 4 + 3] = (__bf16)rw[i].w;
        }
        *(v8bf*)&As[0][srow][skc]     = pa[0];
        *(v8bf*)&As[0][srow][skc + 8] = pa[1];
        *(v8bf*)&Bs[0][srow][skc]     = pw[0];
        *(v8bf*)&Bs[0][srow][skc + 8] = pw[1];
    }
    __syncthreads();

    // fragment addressing (this code's contiguous k = 16*(lane/16) + s for both A and B)
    const int mrow = lane & 15;
    const int kb   = (lane >> 4) * 16;    // A: K-half offset
    const int kh   = (lane >> 4) * 16;    // B: K-half offset

    int buf = 0;
    for (int k0 = 0; k0 < K; k0 += 32) {
        const bool has_next = (k0 + 32) < K;

        // issue ALL next-tile global loads up-front (independent registers)
        float4 ra[4], rw[4];
        if (has_next) {
            const float* An = Arow + (k0 + 32);
            const float* Wn = Wrow + (k0 + 32);
            #pragma unroll
            for (int i = 0; i < 4; ++i) ra[i] = *(const float4*)(An + i * 4);
            #pragma unroll
            for (int i = 0; i < 4; ++i) rw[i] = *(const float4*)(Wn + i * 4);
        }

        // compute on current LDS buffer
        v16bf afrag[2], bfrag[2];
        #pragma unroll
        for (int i = 0; i < 2; ++i) {
            const int m = wm + i * 16 + mrow;
            const v8bf lo = *(const v8bf*)&As[buf][m][kb];
            const v8bf hi = *(const v8bf*)&As[buf][m][kb + 8];
            v16bf a;
            #pragma unroll
            for (int e = 0; e < 8; ++e) { a[e] = lo[e]; a[8 + e] = hi[e]; }
            afrag[i] = a;
            const int n = wn + i * 16 + mrow;
            bfrag[i] = *(const v16bf*)&Bs[buf][n][kh];
        }
        #pragma unroll
        for (int i = 0; i < 2; ++i)
            #pragma unroll
            for (int j = 0; j < 2; ++j)
                acc[i][j] = __builtin_amdgcn_wmma_f32_16x16x32_bf16(
                    false, afrag[i], false, bfrag[j], (short)0, acc[i][j], false, false);

        // convert + store next tile into the other buffer
        if (has_next) {
            v8bf pa[2], pw[2];
            #pragma unroll
            for (int i = 0; i < 4; ++i) {
                pa[i >> 1][(i & 1) * 4 + 0] = (__bf16)ra[i].x;
                pa[i >> 1][(i & 1) * 4 + 1] = (__bf16)ra[i].y;
                pa[i >> 1][(i & 1) * 4 + 2] = (__bf16)ra[i].z;
                pa[i >> 1][(i & 1) * 4 + 3] = (__bf16)ra[i].w;
                pw[i >> 1][(i & 1) * 4 + 0] = (__bf16)rw[i].x;
                pw[i >> 1][(i & 1) * 4 + 1] = (__bf16)rw[i].y;
                pw[i >> 1][(i & 1) * 4 + 2] = (__bf16)rw[i].z;
                pw[i >> 1][(i & 1) * 4 + 3] = (__bf16)rw[i].w;
            }
            *(v8bf*)&As[buf ^ 1][srow][skc]     = pa[0];
            *(v8bf*)&As[buf ^ 1][srow][skc + 8] = pa[1];
            *(v8bf*)&Bs[buf ^ 1][srow][skc]     = pw[0];
            *(v8bf*)&Bs[buf ^ 1][srow][skc + 8] = pw[1];
        }
        __syncthreads();
        buf ^= 1;
    }

    // C/D layout: VGPR r, lanes 0-15 -> M=r, lanes 16-31 -> M=r+8; N = lane&15
    const int lrow = (lane >> 4) * 8;
    const int lcol = lane & 15;
    #pragma unroll
    for (int i = 0; i < 2; ++i)
        #pragma unroll
        for (int j = 0; j < 2; ++j)
            #pragma unroll
            for (int r = 0; r < 8; ++r)
                C[(size_t)(tileM + wm + i * 16 + r + lrow) * N + (tileN + wn + j * 16 + lcol)]
                    = acc[i][j][r];
}

// ---------------------------------------------------------------------------
// Skinny projections: beta = sigmoid(h@W_b^T), g = -exp(A_log)*softplus(h@W_a^T + dt_bias)
// One wave per row; lanes 0-15 do W_b heads, lanes 16-31 do W_a heads.
// ---------------------------------------------------------------------------
__global__ __launch_bounds__(32)
void gdn_proj_small(const float* __restrict__ hs,
                    const float* __restrict__ W_b, const float* __restrict__ W_a,
                    const float* __restrict__ dt_bias, const float* __restrict__ A_log,
                    float* __restrict__ bbuf, float* __restrict__ gbuf) {
    const int m    = blockIdx.x;          // row in [0, B*S)
    const int lane = threadIdx.x;
    const bool isA = lane >= 16;
    const int  h   = lane & 15;
    const float* w = (isA ? W_a : W_b) + (size_t)h * H_;
    const float* x = hs + (size_t)m * H_;
    float acc = 0.f;
    for (int k = 0; k < H_; k += 4) {
        const float4 xv = *(const float4*)&x[k];
        const float4 wv = *(const float4*)&w[k];
        acc = fmaf(xv.x, wv.x, acc);
        acc = fmaf(xv.y, wv.y, acc);
        acc = fmaf(xv.z, wv.z, acc);
        acc = fmaf(xv.w, wv.w, acc);
    }
    if (isA) {
        const float t  = acc + dt_bias[h];
        const float sp = (t > 20.f) ? t : log1pf(expf(t));
        gbuf[(size_t)m * NVH_ + h] = -expf(A_log[h]) * sp;
    } else {
        bbuf[(size_t)m * NVH_ + h] = 1.f / (1.f + expf(-acc));
    }
}

// ---------------------------------------------------------------------------
// Gated delta-rule scan. One block per (b, value-head); 64 threads, thread d
// owns state column S[:, d] in 64 VGPRs. Depthwise causal conv (KSZ=4) + SiLU
// inline via a 4-deep register ring. Prefetch next timestep's channels to
// hide latency on the serial critical path.
// ---------------------------------------------------------------------------
__global__ __launch_bounds__(64)
void gdn_recurrence(const float* __restrict__ mixedRaw, const float* __restrict__ conv_w,
                    const float* __restrict__ gbuf, const float* __restrict__ bbuf,
                    float* __restrict__ obuf) {
    const int bh = blockIdx.x;
    const int b  = bh / NVH_;
    const int h  = bh % NVH_;
    const int kh = h >> 1;                       // GQA: rep = NVH/NKH = 2
    const int d  = threadIdx.x;                  // 0..63

    const int cq = kh * DK_ + d;                 // channel indices in CONV_DIM
    const int ck = KEY_DIM_ + kh * DK_ + d;
    const int cv = 2 * KEY_DIM_ + h * DV_ + d;

    float wq[4], wk[4], wv[4];
    #pragma unroll
    for (int j = 0; j < 4; ++j) {
        wq[j] = conv_w[cq * 4 + j];
        wk[j] = conv_w[ck * 4 + j];
        wv[j] = conv_w[cv * 4 + j];
    }
    float rq[4] = {0.f, 0.f, 0.f, 0.f};
    float rk[4] = {0.f, 0.f, 0.f, 0.f};
    float rv[4] = {0.f, 0.f, 0.f, 0.f};

    float st[64];
    #pragma unroll
    for (int k = 0; k < 64; ++k) st[k] = 0.f;

    __shared__ float qs[64];
    __shared__ float ks[64];

    const float* base = mixedRaw + (size_t)b * S_ * CONV_DIM_;

    for (int t = 0; t < S_; ++t) {
        const int r = t & 3;
        const float* row = base + (size_t)t * CONV_DIM_;
        rq[r] = row[cq];  rk[r] = row[ck];  rv[r] = row[cv];

        if (t + 1 < S_) {                        // hide next step's loads
            __builtin_prefetch(row + CONV_DIM_ + cq, 0, 0);
            __builtin_prefetch(row + CONV_DIM_ + ck, 0, 0);
            __builtin_prefetch(row + CONV_DIM_ + cv, 0, 0);
        }

        float yq = 0.f, yk = 0.f, yv = 0.f;
        #pragma unroll
        for (int j = 0; j < 4; ++j) {                 // tap j -> time t-3+j
            const int slot = (t + 1 + j) & 3;
            yq = fmaf(wq[j], rq[slot], yq);
            yk = fmaf(wk[j], rk[slot], yk);
            yv = fmaf(wv[j], rv[slot], yv);
        }
        yq = yq / (1.f + expf(-yq)) * 0.125f;         // silu + DK^-0.5
        yk = yk / (1.f + expf(-yk));
        yv = yv / (1.f + expf(-yv));

        qs[d] = yq;  ks[d] = yk;
        __syncthreads();

        const float g    = gbuf[((size_t)b * S_ + t) * NVH_ + h];
        const float beta = bbuf[((size_t)b * S_ + t) * NVH_ + h];
        const float ge   = expf(g);

        float dot = 0.f;
        #pragma unroll 8
        for (int k = 0; k < 64; ++k) dot = fmaf(st[k], ks[k], dot);
        const float delta = (yv - ge * dot) * beta;

        float o = 0.f;
        #pragma unroll 8
        for (int k = 0; k < 64; ++k) {
            st[k] = fmaf(st[k], ge, ks[k] * delta);
            o     = fmaf(st[k], qs[k], o);
        }
        obuf[(((size_t)b * S_ + t) * NVH_ + h) * DV_ + d] = o;
        __syncthreads();
    }
}

// ---------------------------------------------------------------------------
// Gated RMSNorm over DV: h = o * silu(z); h *= rsqrt(mean(h^2)+eps) * weight
// ---------------------------------------------------------------------------
__global__ __launch_bounds__(256)
void gdn_gated_rmsnorm(const float* __restrict__ obuf, const float* __restrict__ zbuf,
                       const float* __restrict__ nw, float* __restrict__ hnorm, int total) {
    const int i = blockIdx.x * blockDim.x + threadIdx.x;
    if (i >= total) return;
    const float* o = obuf + (size_t)i * DV_;
    const float* z = zbuf + (size_t)i * DV_;
    float tmp[DV_];
    float sq = 0.f;
    #pragma unroll 8
    for (int dv = 0; dv < DV_; ++dv) {
        const float zv = z[dv];
        const float hv = o[dv] * (zv / (1.f + expf(-zv)));
        tmp[dv] = hv;
        sq = fmaf(hv, hv, sq);
    }
    const float inv = rsqrtf(sq * (1.f / DV_) + 1e-6f);
    #pragma unroll 8
    for (int dv = 0; dv < DV_; ++dv)
        hnorm[(size_t)i * DV_ + dv] = tmp[dv] * inv * nw[dv];
}

// ---------------------------------------------------------------------------
extern "C" void kernel_launch(void* const* d_in, const int* in_sizes, int n_in,
                              void* d_out, int out_size, void* d_ws, size_t ws_size,
                              hipStream_t stream) {
    const float* hs      = (const float*)d_in[0];
    const float* W_qkv   = (const float*)d_in[1];
    const float* conv_w  = (const float*)d_in[2];
    const float* W_z     = (const float*)d_in[3];
    const float* W_b     = (const float*)d_in[4];
    const float* W_a     = (const float*)d_in[5];
    const float* dt_bias = (const float*)d_in[6];
    const float* A_log   = (const float*)d_in[7];
    const float* norm_w  = (const float*)d_in[8];
    const float* W_out   = (const float*)d_in[9];
    float* out = (float*)d_out;

    const int M = B_ * S_;                         // 8192 rows

    float* ws = (float*)d_ws;
    size_t off = 0;
    float* mixedRaw = ws + off;  off += (size_t)M * CONV_DIM_;   // 64 MB
    float* zbuf     = ws + off;  off += (size_t)M * VAL_DIM_;    // 32 MB
    float* gbuf     = ws + off;  off += (size_t)M * NVH_;
    float* bbuf     = ws + off;  off += (size_t)M * NVH_;
    float* obuf     = ws + off;  off += (size_t)M * VAL_DIM_;    // 32 MB
    float* hnorm    = ws + off;  off += (size_t)M * VAL_DIM_;    // 32 MB

    // 1) fused QKV projection (pre-conv): mixedRaw = hs @ W_qkv^T
    gdn_wmma_gemm_xWT<<<dim3(CONV_DIM_ / 64, M / 64), 128, 0, stream>>>(
        hs, W_qkv, mixedRaw, M, CONV_DIM_, H_);

    // 2) z projection: zbuf = hs @ W_z^T
    gdn_wmma_gemm_xWT<<<dim3(VAL_DIM_ / 64, M / 64), 128, 0, stream>>>(
        hs, W_z, zbuf, M, VAL_DIM_, H_);

    // 3) skinny beta / gate projections
    gdn_proj_small<<<M, 32, 0, stream>>>(hs, W_b, W_a, dt_bias, A_log, bbuf, gbuf);

    // 4) conv+silu fused gated delta-rule scan
    gdn_recurrence<<<B_ * NVH_, 64, 0, stream>>>(mixedRaw, conv_w, gbuf, bbuf, obuf);

    // 5) gated RMSNorm
    const int total = M * NVH_;                    // 131072 (b,s,h) rows
    gdn_gated_rmsnorm<<<(total + 255) / 256, 256, 0, stream>>>(obuf, zbuf, norm_w, hnorm, total);

    // 6) output projection: out = hnorm @ W_out^T
    gdn_wmma_gemm_xWT<<<dim3(H_ / 64, M / 64), 128, 0, stream>>>(
        hnorm, W_out, out, M, H_, VAL_DIM_);
}
